// Fastformer_3487513445000
// MI455X (gfx1250) — matrix operand order repaired and checked by reading, not compiled, not run on hardware
//
#include <hip/hip_runtime.h>

// ---------------------------------------------------------------------------
// Fastformer (B=8, S=2048, D=1024, h=16, dh=64) for MI455X / gfx1250.
// Big GEMMs in bf16 WMMA (v_wmma_f32_16x16x32_bf16), reductions in fp32.
// B-tile staging: MODE 0/1 via GLOBAL_LOAD_ASYNC_TO_LDS_B128 (ASYNCcnt),
// MODE 2 via TDM TENSOR_LOAD_TO_LDS with LDS padding (TENSORcnt).
// A-tiles converted f32->bf16 through VGPRs into LDS.
// ---------------------------------------------------------------------------

typedef __attribute__((ext_vector_type(16))) __bf16 v16bf;
typedef __attribute__((ext_vector_type(8)))  float  v8f;
typedef __attribute__((ext_vector_type(4)))  unsigned int v4u;
typedef __attribute__((ext_vector_type(8)))  int  v8i;
typedef __attribute__((ext_vector_type(4)))  int  v4i;

#define D_MODEL 1024
#define SEQ     2048
#define BATCH   8
#define NROWS   (BATCH * SEQ)      // 16384
#define AP      40                 // LDS row pitch in bf16 (32 + 8 pad -> 80B)

static __device__ __forceinline__ unsigned short f2bf(float f) {
  unsigned int u = __float_as_uint(f);
  u += 0x7fffu + ((u >> 16) & 1u);          // round-to-nearest-even
  return (unsigned short)(u >> 16);
}
static __device__ __forceinline__ float bf2f(unsigned short h) {
  return __uint_as_float(((unsigned int)h) << 16);
}
static __device__ __forceinline__ unsigned int pk2(float a, float b) {
  return (unsigned int)f2bf(a) | ((unsigned int)f2bf(b) << 16);
}

// Low 32 bits of a generic pointer into __shared__ = LDS byte offset
// (see ISA 10.2, LDS aperture).
static __device__ __forceinline__ unsigned lds_off(const void* p) {
  return (unsigned)(unsigned long long)(p);
}

// Async global -> LDS, 16 bytes per lane, tracked by ASYNCcnt.
static __device__ __forceinline__ void async_g2l_b128(unsigned lds_addr,
                                                      const void* gaddr) {
  asm volatile("global_load_async_to_lds_b128 %0, %1, off"
               :
               : "v"(lds_addr), "v"((unsigned long long)gaddr)
               : "memory");
}
static __device__ __forceinline__ void wait_asynccnt0() {
  asm volatile("s_wait_asynccnt 0x0" ::: "memory");
}

// TDM: load a [rows=128][K=32 bf16] tile (row stride 1024 elems) into LDS at
// lds_addr with 16B padding after each 64B row (-> 80B pitch). ISA 08 §8.
static __device__ __forceinline__ void tdm_load_b_tile(unsigned lds_addr,
                                                       const void* gaddr) {
  const unsigned long long ga = (unsigned long long)gaddr;
  v4u g0; v8i g1; v4i g2; v4i g3;
  g0[0] = 1u;                                   // count=1, user descriptor
  g0[1] = lds_addr;                             // lds_addr[31:0]
  g0[2] = (unsigned)ga;                         // global_addr[31:0]
  g0[3] = (unsigned)((ga >> 32) & 0x01FFFFFFu)  // global_addr[56:32]
        | 0x80000000u;                          // type=2 ("image")
  g1[0] = (int)((1u << 16)                      // data_size = 2 bytes
        |       (1u << 20)                      // pad_enable
        |       (3u << 22)                      // pad_interval: 16 DWORDs (64B)
        |       (3u << 25));                    // pad_amount:   4 DWORDs (16B)
  g1[1] = (int)(1024u << 16);                   // tensor_dim0 = 1024 (lo16)
  g1[2] = (int)(1024u << 16);                   // dim0 hi=0 | tensor_dim1 lo16
  g1[3] = (int)(32u << 16);                     // dim1 hi=0 | tile_dim0 = 32
  g1[4] = 128;                                  // tile_dim1 = 128, tile_dim2 = 0
  g1[5] = 1024;                                 // tensor_dim0_stride lo (elems)
  g1[6] = 0;                                    // stride hi | dim1_stride lo
  g1[7] = 0;
  g2[0] = g2[1] = g2[2] = g2[3] = 0;            // <=2D: groups 2/3 unused
  g3[0] = g3[1] = g3[2] = g3[3] = 0;
#if defined(__clang_major__) && (__clang_major__ >= 23)
  v8i g4;
  g4[0] = g4[1] = g4[2] = g4[3] = g4[4] = g4[5] = g4[6] = g4[7] = 0;
  __builtin_amdgcn_tensor_load_to_lds(g0, g1, g2, g3, g4, 0);
#else
  __builtin_amdgcn_tensor_load_to_lds(g0, g1, g2, g3, 0);
#endif
}

// A fragment: 16x32 bf16, lanes0-15: K0..7 + K16..23 ; lanes16-31: K8..15 + K24..31
static __device__ __forceinline__ v16bf frag_a(const unsigned short* base, int lane) {
  const int row = lane & 15, sub = lane >> 4;
  const unsigned short* p = base + row * AP;
  union { uint4 u[2]; v16bf v; } x;
  x.u[0] = *(const uint4*)(p + sub * 8);
  x.u[1] = *(const uint4*)(p + 16 + sub * 8);
  return x.v;
}
// B fragment from N-major LDS tile: lanes0-15: K0..7 + K16..23 ; lanes16-31: K8..15 + K24..31
static __device__ __forceinline__ v16bf frag_b(const unsigned short* base, int lane) {
  const int col = lane & 15, sub = lane >> 4;
  const unsigned short* p = base + col * AP + sub * 8;
  union { uint4 u[2]; v16bf v; } x;
  x.u[0] = *(const uint4*)(p);
  x.u[1] = *(const uint4*)(p + 16);
  return x.v;
}

// ---------------------------------------------------------------------------
// WMMA GEMM: C[M,1024] = A[M,1024](f32 -> bf16) * BT[1024,1024](bf16, [N][K])
// MODE 0: C = A @ W                       (Q projection)
// MODE 1: scrambled store QAK[b,h,s,j] = (A@W)[b,s,h*64+j] * qg[b, h*64+j]
// MODE 2: per-batch B (grid.z), C = A @ Wb + Qres   (output proj + residual)
// ---------------------------------------------------------------------------
template <int MODE>
__global__ __launch_bounds__(256)
void gemm128_wmma(const float* __restrict__ A, const unsigned short* __restrict__ BT,
                  float* __restrict__ C, const float* __restrict__ qg,
                  const float* __restrict__ Qres) {
  __shared__ unsigned short As[128 * AP];
  __shared__ unsigned short Bs[128 * AP];

  const int t    = threadIdx.x;
  const int lane = t & 31;
  const int wid  = t >> 5;
  const int wm   = wid >> 2;          // 0..1 (64 rows each)
  const int wn   = wid & 3;           // 0..3 (32 cols each)

  const int  bz      = (MODE == 2) ? blockIdx.z : 0;
  const long rowBase = (long)bz * SEQ + (long)blockIdx.y * 128;
  const int  colBase = blockIdx.x * 128;
  const unsigned short* Bp = BT + ((MODE == 2) ? (size_t)bz * D_MODEL * D_MODEL : 0);

  v8f acc[4][2] = {};

  const int arow = t >> 3;            // 0..31
  const int acol = (t & 7) * 4;       // 0..28

  // Per-thread B staging coordinates (2 x 16B chunks per K-step, modes 0/1)
  const int brow0 = t >> 2;                 // 0..63
  const int bcol0 = (t & 3) * 8;            // 0,8,16,24
  const unsigned bl0 = lds_off(&Bs[brow0 * AP + bcol0]);
  const unsigned bl1 = lds_off(&Bs[(brow0 + 64) * AP + bcol0]);
  const unsigned bsBase = lds_off(&Bs[0]);

  for (int kt = 0; kt < 32; ++kt) {
    const int k0 = kt * 32;

    // --- stage B tile [N=128][K=32] into LDS (no VGPR data path) ---
    if constexpr (MODE == 2) {
      if (wid == 0) {   // one wave drives the Tensor Data Mover
        tdm_load_b_tile(bsBase, Bp + (size_t)colBase * D_MODEL + k0);
      }
    } else {
      async_g2l_b128(bl0, Bp + (size_t)(colBase + brow0) * D_MODEL + k0 + bcol0);
      async_g2l_b128(bl1, Bp + (size_t)(colBase + brow0 + 64) * D_MODEL + k0 + bcol0);
    }

    // --- stage A tile (f32 -> bf16), 128x32 ---
#pragma unroll
    for (int rr = 0; rr < 4; ++rr) {
      const int row = arow + rr * 32;
      const float4 f = *(const float4*)(A + (size_t)(rowBase + row) * D_MODEL + k0 + acol);
      uint2 u;
      u.x = pk2(f.x, f.y);
      u.y = pk2(f.z, f.w);
      *(uint2*)(&As[row * AP + acol]) = u;
    }
    if (kt + 1 < 32)  // hint next A tile into cache (global_prefetch_b8)
      __builtin_prefetch(A + (size_t)(rowBase + arow) * D_MODEL + k0 + 32 + acol, 0, 0);

    // --- B tile landed in LDS before the barrier ---
    if constexpr (MODE == 2) {
      if (wid == 0) __builtin_amdgcn_s_wait_tensorcnt((short)0);
    } else {
      wait_asynccnt0();
    }
    __syncthreads();

    v16bf af[4], bfv[2];
#pragma unroll
    for (int i = 0; i < 4; ++i) af[i] = frag_a(&As[(wm * 64 + i * 16) * AP], lane);
#pragma unroll
    for (int j = 0; j < 2; ++j) bfv[j] = frag_b(&Bs[(wn * 32 + j * 16) * AP], lane);

#pragma unroll
    for (int i = 0; i < 4; ++i)
#pragma unroll
      for (int j = 0; j < 2; ++j)
        acc[i][j] = __builtin_amdgcn_wmma_f32_16x16x32_bf16(
            false, af[i], false, bfv[j], (short)0, acc[i][j], false, false);
    __syncthreads();
  }

  // --- epilogue ---
#pragma unroll
  for (int i = 0; i < 4; ++i) {
#pragma unroll
    for (int j = 0; j < 2; ++j) {
#pragma unroll
      for (int r = 0; r < 8; ++r) {
        const int  row_l = wm * 64 + i * 16 + r + ((lane >> 4) << 3);
        const int  col_l = wn * 32 + j * 16 + (lane & 15);
        const long row_g = rowBase + row_l;
        const int  col_g = colBase + col_l;
        float v = acc[i][j][r];
        if (MODE == 0) {
          C[(size_t)row_g * D_MODEL + col_g] = v;
        } else if (MODE == 1) {
          const long b = row_g >> 11, s = row_g & (SEQ - 1);
          v *= qg[b * D_MODEL + col_g];
          const int h = col_g >> 6, jj = col_g & 63;
          C[(((size_t)b * 16 + h) * SEQ + s) * 64 + jj] = v;
        } else {
          const size_t idx = (size_t)row_g * D_MODEL + col_g;
          C[idx] = v + Qres[idx];
        }
      }
    }
  }
}

// ---------------------------------------------------------------------------
// W[k][n] f32  ->  WT[n][k] bf16   (1024x1024)
// ---------------------------------------------------------------------------
__global__ __launch_bounds__(256)
void k_wtrans(const float* __restrict__ W, unsigned short* __restrict__ WT) {
  const int idx = blockIdx.x * 256 + threadIdx.x;    // n*1024 + k
  const int n = idx >> 10, k = idx & 1023;
  WT[idx] = f2bf(W[k * D_MODEL + n]);
}

// ---------------------------------------------------------------------------
// logits[b,e,s] = 0.125 * X[row=b*S+s, :] . W[:, e]   (W is [1024][16] f32)
// block = 4 rows x 16 e x 4 partials
// ---------------------------------------------------------------------------
__global__ __launch_bounds__(256)
void k_nk16(const float* __restrict__ X, const float* __restrict__ W,
            float* __restrict__ L) {
  __shared__ float red[256];
  const int t = threadIdx.x;
  const int o = t & 63, p = t >> 6;
  const int r = o >> 4, e = o & 15;
  const size_t rid = (size_t)blockIdx.x * 4 + r;
  const float* row = X + rid * D_MODEL;
  const float* wc  = W + e;
  float s = 0.f;
  const int k0 = p * 256;
  for (int k = k0; k < k0 + 256; ++k) s += row[k] * wc[k * 16];
  red[t] = s;
  __syncthreads();
  if (t < 64) {
    const float v = (red[t] + red[t + 64] + red[t + 128] + red[t + 192]) * 0.125f;
    const size_t b = rid >> 11, ss = rid & (SEQ - 1);
    L[(b * 16 + e) * SEQ + ss] = v;
  }
}

// ---------------------------------------------------------------------------
// Per (b,h): softmax over 2048 logits, then OUT[b,1024] = weighted pool of X.
// X element (s, j) at: X + b*bStride + h*hStride + s*sStride + j   (j = 0..63)
// ---------------------------------------------------------------------------
__global__ __launch_bounds__(256)
void k_softmax_pool(const float* __restrict__ L, const float* __restrict__ X,
                    size_t bStride, size_t hStride, size_t sStride,
                    float* __restrict__ OUT) {
  __shared__ float w[SEQ];
  __shared__ float red[256];
  const int t = threadIdx.x;
  const int g = blockIdx.x, b = g >> 4, h = g & 15;
  const float* lg = L + (size_t)g * SEQ;

  float m = -3.4e38f;
  for (int s = t; s < SEQ; s += 256) { const float v = lg[s]; w[s] = v; m = fmaxf(m, v); }
  red[t] = m; __syncthreads();
  for (int o = 128; o > 0; o >>= 1) { if (t < o) red[t] = fmaxf(red[t], red[t + o]); __syncthreads(); }
  const float M = red[0]; __syncthreads();

  float e = 0.f;
  for (int s = t; s < SEQ; s += 256) { const float ex = __expf(w[s] - M); w[s] = ex; e += ex; }
  red[t] = e; __syncthreads();
  for (int o = 128; o > 0; o >>= 1) { if (t < o) red[t] += red[t + o]; __syncthreads(); }
  const float inv = 1.0f / red[0]; __syncthreads();

  const int j = t & 63, p = t >> 6;
  const float* xb = X + (size_t)b * bStride + (size_t)h * hStride + j;
  float acc = 0.f;
  for (int s = p; s < SEQ; s += 4) acc += w[s] * xb[(size_t)s * sStride];
  red[t] = acc; __syncthreads();
  if (t < 64) {
    OUT[(size_t)b * D_MODEL + h * 64 + t] =
        (red[t] + red[t + 64] + red[t + 128] + red[t + 192]) * inv;
  }
}

// WPbT[b][n][k] = WPT[n][k] * kg[b][k]   (bf16 out)
__global__ __launch_bounds__(256)
void k_scale_wpt(const unsigned short* __restrict__ WPT, const float* __restrict__ kg,
                 unsigned short* __restrict__ OUT) {
  const size_t gid = (size_t)blockIdx.x * 256 + threadIdx.x;  // 8M
  const size_t b = gid >> 20, nk = gid & ((1u << 20) - 1);
  const int k = (int)(nk & 1023);
  OUT[gid] = f2bf(bf2f(WPT[nk]) * kg[b * D_MODEL + k]);
}

// ---------------------------------------------------------------------------
extern "C" void kernel_launch(void* const* d_in, const int* in_sizes, int n_in,
                              void* d_out, int out_size, void* d_ws, size_t ws_size,
                              hipStream_t stream) {
  const float* Q_seq = (const float*)d_in[0];
  const float* K_seq = (const float*)d_in[1];
  // d_in[2] = V_seq : unused (faithful to reference)
  const float* WQ = (const float*)d_in[3];
  const float* WK = (const float*)d_in[4];
  const float* Wa = (const float*)d_in[5];
  const float* Wb = (const float*)d_in[6];
  const float* WP = (const float*)d_in[7];
  float* out = (float*)d_out;

  char* ws = (char*)d_ws;
  const size_t SZ_Q   = (size_t)NROWS * D_MODEL * 4;          // 64 MB
  const size_t SZ_WT  = (size_t)D_MODEL * D_MODEL * 2;        // 2 MB
  const size_t SZ_LOG = (size_t)BATCH * 16 * SEQ * 4;         // 1 MB
  float*          Qf   = (float*)(ws);                         // Q = Q_seq@WQ
  float*          QAK  = (float*)(ws + SZ_Q);                  // scrambled [b,h,s,j]
  unsigned short* WQT  = (unsigned short*)(ws + 2 * SZ_Q);
  unsigned short* WKT  = (unsigned short*)(ws + 2 * SZ_Q + SZ_WT);
  unsigned short* WPT  = (unsigned short*)(ws + 2 * SZ_Q + 2 * SZ_WT);
  unsigned short* WPBT = (unsigned short*)(ws + 2 * SZ_Q + 3 * SZ_WT);   // 8x2MB
  float*          LA   = (float*)(ws + 2 * SZ_Q + 3 * SZ_WT + 8 * SZ_WT);
  float*          LB   = (float*)((char*)LA + SZ_LOG);
  float*          qg   = (float*)((char*)LB + SZ_LOG);
  float*          kg   = (float*)((char*)qg + BATCH * D_MODEL * 4);

  const dim3 blk(256);

  // 1) weight transpose + bf16 convert
  k_wtrans<<<4096, blk, 0, stream>>>(WQ, WQT);
  k_wtrans<<<4096, blk, 0, stream>>>(WK, WKT);
  k_wtrans<<<4096, blk, 0, stream>>>(WP, WPT);

  // 2) Q = Q_seq @ WQ
  gemm128_wmma<0><<<dim3(8, 128, 1), blk, 0, stream>>>(Q_seq, WQT, Qf, nullptr, nullptr);

  // 3) alpha logits + softmax + q_glob pooling
  k_nk16<<<NROWS / 4, blk, 0, stream>>>(Qf, Wa, LA);
  k_softmax_pool<<<128, blk, 0, stream>>>(LA, Qf, (size_t)SEQ * D_MODEL, 64, D_MODEL, qg);

  // 4) QAK = (K_seq @ WK) * q_glob  -> scrambled [b,h,s,j]
  gemm128_wmma<1><<<dim3(8, 128, 1), blk, 0, stream>>>(K_seq, WKT, QAK, qg, nullptr);

  // 5) beta logits (QAK flat view) + softmax + k_glob pooling
  k_nk16<<<NROWS / 4, blk, 0, stream>>>(QAK, Wb, LB);
  k_softmax_pool<<<128, blk, 0, stream>>>(LB, QAK, (size_t)16 * SEQ * 64,
                                          (size_t)SEQ * 64, 64, kg);

  // 6) per-batch WPb = diag(kg) @ WP, then out = Q @ WPb + Q
  k_scale_wpt<<<BATCH * 4096, blk, 0, stream>>>(WPT, kg, WPBT);
  gemm128_wmma<2><<<dim3(8, 16, BATCH), blk, 0, stream>>>(Qf, WPBT, out, nullptr, Qf);
}
